// WeightedScaledDotProductAttention_86174223827276
// MI455X (gfx1250) — matrix operand order repaired and checked by reading, not compiled, not run on hardware
//
#include <hip/hip_runtime.h>

typedef __attribute__((ext_vector_type(16))) _Float16 v16h;
typedef __attribute__((ext_vector_type(8)))  _Float16 v8h;
typedef __attribute__((ext_vector_type(4)))  _Float16 v4h;
typedef __attribute__((ext_vector_type(8)))  float    v8f;
typedef __attribute__((ext_vector_type(4)))  float    v4f;

#define BATCH 8
#define SEQ   2048
#define DIM   1024

// ---------------------------------------------------------------------------
// Fragment loaders (layouts per CDNA5 ISA 7.12.2, wave32):
//  A 16-bit 16x32 (MxK): lane m (0-15): elems 0-7 = K 0-7, elems 8-15 = K 16-23
//                        lane m+16    : elems 0-7 = K 8-15, elems 8-15 = K 24-31
//  B 16-bit 32x16 (KxN): lane n (0-15): elems 0-7 = K 0-7, elems 8-15 = K 16-23
//                        lane n+16    : elems 0-7 = K 8-15, elems 8-15 = K 24-31
// ---------------------------------------------------------------------------

__device__ __forceinline__ v16h load_a_f16(const _Float16* __restrict__ base,
                                           int ld, int rowBase, int k0, int lane) {
  const int half = lane >> 4;
  const int m    = lane & 15;
  const _Float16* p = base + (size_t)(rowBase + m) * ld + k0 + 8 * half;
  v8h lo = *(const v8h*)(p);        // K = k0 + 8*half + [0..7]
  v8h hi = *(const v8h*)(p + 16);   // K = k0 + 16 + 8*half + [0..7]
  v16h a;
#pragma unroll
  for (int e = 0; e < 8; ++e) { a[e] = lo[e]; a[8 + e] = hi[e]; }
  return a;
}

__device__ __forceinline__ v16h load_a_f32cvt(const float* __restrict__ base,
                                              int ld, int rowBase, int k0, int lane) {
  const int half = lane >> 4;
  const int m    = lane & 15;
  const float* p = base + (size_t)(rowBase + m) * ld + k0 + 8 * half;
  v16h a;
#pragma unroll
  for (int e = 0; e < 8; ++e) {
    a[e]     = (_Float16)p[e];
    a[8 + e] = (_Float16)p[16 + e];
  }
  return a;
}

__device__ __forceinline__ v16h load_b_f16(const _Float16* __restrict__ base,
                                           int ld, int colBase, int k0, int lane) {
  const int half = lane >> 4;
  const int n    = lane & 15;
  const _Float16* p = base + (size_t)(colBase + n) * ld + k0 + 8 * half;
  v8h lo = *(const v8h*)(p);        // K = k0 + 8*half + [0..7]
  v8h hi = *(const v8h*)(p + 16);   // K = k0 + 16 + 8*half + [0..7]
  v16h b;
#pragma unroll
  for (int e = 0; e < 8; ++e) { b[e] = lo[e]; b[8 + e] = hi[e]; }
  return b;
}

// ---------------------------------------------------------------------------
// W1 f32 -> f16 conversion (once per launch; 1M elements)
// ---------------------------------------------------------------------------
__global__ __launch_bounds__(256) void cvt_w1_kernel(const float* __restrict__ w1,
                                                     _Float16* __restrict__ w1h) {
  const int i = (blockIdx.x * 256 + threadIdx.x) * 4;
  v4f x = *(const v4f*)(w1 + i);
  v4h y;
  y[0] = (_Float16)x[0]; y[1] = (_Float16)x[1];
  y[2] = (_Float16)x[2]; y[3] = (_Float16)x[3];
  *(v4h*)(w1h + i) = y;
}

// ---------------------------------------------------------------------------
// GEMM1: q_out[r,e] = sum_d q[r,d] * W1[e,d] + b1[e]   (M=16384, N=1024, K=1024)
// block = 256 threads (8 waves), 128x256 tile; wave = 64x64 tile (4x4 WMMA subtiles)
// ---------------------------------------------------------------------------
__global__ __launch_bounds__(256) void linear_gemm_kernel(
    const float* __restrict__ q, const _Float16* __restrict__ w1h,
    const float* __restrict__ b1, _Float16* __restrict__ q_out) {
  const int lane = threadIdx.x & 31;
  const int wave = threadIdx.x >> 5;
  const int rowBase = blockIdx.x * 128 + (wave >> 2) * 64;   // 2 row groups
  const int colBase = blockIdx.y * 256 + (wave & 3) * 64;    // 4 col groups

  const v8f zero = {};
  v8f acc[4][4];
#pragma unroll
  for (int i = 0; i < 4; ++i)
#pragma unroll
    for (int j = 0; j < 4; ++j) acc[i][j] = zero;

#pragma unroll 1
  for (int k0 = 0; k0 < DIM; k0 += 32) {
    v16h af[4];
#pragma unroll
    for (int i = 0; i < 4; ++i)
      af[i] = load_a_f32cvt(q, DIM, rowBase + 16 * i, k0, lane);
    v16h bf[4];
#pragma unroll
    for (int j = 0; j < 4; ++j)
      bf[j] = load_b_f16(w1h, DIM, colBase + 16 * j, k0, lane);
#pragma unroll
    for (int i = 0; i < 4; ++i)
#pragma unroll
      for (int j = 0; j < 4; ++j)
        acc[i][j] = __builtin_amdgcn_wmma_f32_16x16x32_f16(
            false, af[i], false, bf[j], (short)0, acc[i][j], false, false);
  }

  // epilogue: + bias, convert to f16, store (D layout: elem e -> row 8*half+e, col = lane&15)
  const int half = lane >> 4;
  const int n    = lane & 15;
#pragma unroll
  for (int j = 0; j < 4; ++j) {
    const int c = colBase + 16 * j + n;
    const float bv = b1[c];
#pragma unroll
    for (int i = 0; i < 4; ++i) {
#pragma unroll
      for (int e = 0; e < 8; ++e) {
        const int r = rowBase + 16 * i + 8 * half + e;
        q_out[(size_t)r * DIM + c] = (_Float16)(acc[i][j][e] + bv);
      }
    }
  }
}

// ---------------------------------------------------------------------------
// GEMM2: attn[b,n,m] = sum_e q_out[b,n,e] * q_out[b,m,e]   (per batch 2048x2048x1024)
// block = 256 threads, 128x256 tile; wave = 64x64 tile (4x4 WMMA subtiles)
// ---------------------------------------------------------------------------
__global__ __launch_bounds__(256) void attn_gemm_kernel(
    const _Float16* __restrict__ q_out, float* __restrict__ attn_out) {
  const int b = blockIdx.z;
  const _Float16* Q = q_out + (size_t)b * SEQ * DIM;
  float* attn = attn_out + (size_t)b * SEQ * SEQ;

  const int lane = threadIdx.x & 31;
  const int wave = threadIdx.x >> 5;
  const int rowBase = blockIdx.x * 128 + (wave >> 2) * 64;
  const int colBase = blockIdx.y * 256 + (wave & 3) * 64;

  const v8f zero = {};
  v8f acc[4][4];
#pragma unroll
  for (int i = 0; i < 4; ++i)
#pragma unroll
    for (int j = 0; j < 4; ++j) acc[i][j] = zero;

#pragma unroll 1
  for (int k0 = 0; k0 < DIM; k0 += 32) {
    v16h af[4];
#pragma unroll
    for (int i = 0; i < 4; ++i)
      af[i] = load_a_f16(Q, DIM, rowBase + 16 * i, k0, lane);
    v16h bf[4];
#pragma unroll
    for (int j = 0; j < 4; ++j)
      bf[j] = load_b_f16(Q, DIM, colBase + 16 * j, k0, lane);  // B col n = Q row (colBase+n)
#pragma unroll
    for (int i = 0; i < 4; ++i)
#pragma unroll
      for (int j = 0; j < 4; ++j)
        acc[i][j] = __builtin_amdgcn_wmma_f32_16x16x32_f16(
            false, af[i], false, bf[j], (short)0, acc[i][j], false, false);
  }

  const int half = lane >> 4;
  const int n    = lane & 15;
#pragma unroll
  for (int j = 0; j < 4; ++j) {
    const int c = colBase + 16 * j + n;
#pragma unroll
    for (int i = 0; i < 4; ++i) {
#pragma unroll
      for (int e = 0; e < 8; ++e) {
        const int r = rowBase + 16 * i + 8 * half + e;
        attn[(size_t)r * SEQ + c] = acc[i][j][e];
      }
    }
  }
}

// ---------------------------------------------------------------------------
// rowsum -> d^{-1/2}: one block per (b,n) row of attn
// ---------------------------------------------------------------------------
__global__ __launch_bounds__(256) void rowsum_kernel(const float* __restrict__ attn,
                                                     float* __restrict__ dis) {
  const int row = blockIdx.x;  // 0 .. BATCH*SEQ-1
  const float* p = attn + (size_t)row * SEQ;
  float s = 0.f;
#pragma unroll
  for (int i = threadIdx.x * 4; i < SEQ; i += 256 * 4) {
    v4f x = *(const v4f*)(p + i);
    s += x[0] + x[1] + x[2] + x[3];
  }
  __shared__ float red[256];
  red[threadIdx.x] = s;
  __syncthreads();
#pragma unroll
  for (int off = 128; off > 0; off >>= 1) {
    if (threadIdx.x < off) red[threadIdx.x] += red[threadIdx.x + off];
    __syncthreads();
  }
  if (threadIdx.x == 0) {
    float r = fmaxf(red[0], 1.0f);
    dis[row] = 1.0f / sqrtf(r);
  }
}

// ---------------------------------------------------------------------------
// output[b,n,m] = dis[b,n] * attn[b,n,m] * dis[b,m]
// ---------------------------------------------------------------------------
__global__ __launch_bounds__(256) void scale_kernel(const float* __restrict__ attn,
                                                    const float* __restrict__ dis,
                                                    float* __restrict__ out) {
  const size_t idx = ((size_t)blockIdx.x * 256 + threadIdx.x) * 4;  // flat into [B,N,N]
  const size_t row = idx >> 11;          // b*SEQ + n   (SEQ = 2048)
  const int    col = (int)(idx & (SEQ - 1));
  const size_t bb  = row >> 11;          // batch
  const float dr = dis[row];
  const float* dc = dis + (bb << 11) + col;
  v4f a = *(const v4f*)(attn + idx);
  v4f o;
  o[0] = dr * a[0] * dc[0];
  o[1] = dr * a[1] * dc[1];
  o[2] = dr * a[2] * dc[2];
  o[3] = dr * a[3] * dc[3];
  *(v4f*)(out + idx) = o;
}

// ---------------------------------------------------------------------------
extern "C" void kernel_launch(void* const* d_in, const int* in_sizes, int n_in,
                              void* d_out, int out_size, void* d_ws, size_t ws_size,
                              hipStream_t stream) {
  const float* q  = (const float*)d_in[0];
  // d_in[1] = k (unused by reference), d_in[2] = v (unused by reference)
  const float* W1 = (const float*)d_in[3];
  const float* b1 = (const float*)d_in[4];
  (void)in_sizes; (void)n_in; (void)out_size; (void)ws_size;

  float* out_norm = (float*)d_out;                                   // output (first)
  float* out_attn = (float*)d_out + (size_t)BATCH * SEQ * SEQ;       // attn (second)

  char* ws = (char*)d_ws;
  _Float16* w1h   = (_Float16*)ws;                                               // 2 MB
  _Float16* q_out = (_Float16*)(ws + (size_t)DIM * DIM * sizeof(_Float16));      // 33.5 MB
  float*    dis   = (float*)(ws + (size_t)DIM * DIM * sizeof(_Float16)
                               + (size_t)BATCH * SEQ * DIM * sizeof(_Float16));  // 64 KB

  // 1) W1 -> f16
  cvt_w1_kernel<<<(DIM * DIM) / (256 * 4), 256, 0, stream>>>(W1, w1h);

  // 2) q_out = q @ W1^T + b1 (f16 output)
  dim3 g1((BATCH * SEQ) / 128, DIM / 256);
  linear_gemm_kernel<<<g1, 256, 0, stream>>>(q, w1h, b1, q_out);

  // 3) attn = q_out @ q_out^T (f32, written to second output slot)
  dim3 g2(SEQ / 128, SEQ / 256, BATCH);
  attn_gemm_kernel<<<g2, 256, 0, stream>>>(q_out, out_attn);

  // 4) row sums -> d^{-1/2}
  rowsum_kernel<<<BATCH * SEQ, 256, 0, stream>>>(out_attn, dis);

  // 5) output = D^{-1/2} attn D^{-1/2}
  scale_kernel<<<(unsigned)((size_t)BATCH * SEQ * SEQ / (256 * 4)), 256, 0, stream>>>(
      out_attn, dis, out_norm);
}
